// MultiHeadAttention_80753975099453
// MI455X (gfx1250) — matrix operand order repaired and checked by reading, not compiled, not run on hardware
//
#include <hip/hip_runtime.h>

// ---------------------------------------------------------------------------
// MHA forward for MI455X (gfx1250, wave32, WMMA f16 -> f32 accumulate)
// B=8, S=1024, D_MODEL=1024, H=16, Dh=64
// outputs: out (B,S,1024) fp32  ++  attn (B,H,S,S) fp32   (concatenated)
// ---------------------------------------------------------------------------

typedef __attribute__((ext_vector_type(16))) _Float16 v16h;
typedef __attribute__((ext_vector_type(8)))  _Float16 v8h;
typedef __attribute__((ext_vector_type(8)))  float    v8f;
typedef __attribute__((ext_vector_type(4)))  float    v4f;

#define DM   1024
#define SEQ  1024
#define BB   8
#define NH   16
#define DH   64
#define NEGBIG (-1e9f)

static __device__ __forceinline__ v8f wmma_f16(v16h a, v16h b, v8f c) {
  // (neg_a, A, neg_b, B, c_mod, C, reuse_a, reuse_b)
  return __builtin_amdgcn_wmma_f32_16x16x32_f16(false, a, false, b, (short)0, c,
                                                false, false);
}

static __device__ __forceinline__ v16h cat8(v8h lo, v8h hi) {
  v16h a;
#pragma unroll
  for (int i = 0; i < 8; ++i) { a[i] = lo[i]; a[i + 8] = hi[i]; }
  return a;
}

// A-matrix (16x32 f16) fragment: lanes 0-15 hold K 0..15 of row M=lane,
// lanes 16-31 hold K 16..31 of row M=lane-16 (contiguous, matching B loads).
static __device__ __forceinline__ v16h load_a_f16(const _Float16* row, int kk, int lane) {
  int lo = kk + ((lane & 16) ? 16 : 0);
  int hi = kk + ((lane & 16) ? 24 : 8);
  return cat8(*(const v8h*)(row + lo), *(const v8h*)(row + hi));
}

static __device__ __forceinline__ v16h load_a_f32(const float* row, int kk, int lane) {
  int lo = kk + ((lane & 16) ? 16 : 0);
  int hi = kk + ((lane & 16) ? 24 : 8);
  v4f f0 = *(const v4f*)(row + lo);
  v4f f1 = *(const v4f*)(row + lo + 4);
  v4f f2 = *(const v4f*)(row + hi);
  v4f f3 = *(const v4f*)(row + hi + 4);
  v16h a;
#pragma unroll
  for (int i = 0; i < 4; ++i) {
    a[i]      = (_Float16)f0[i];
    a[i + 4]  = (_Float16)f1[i];
    a[i + 8]  = (_Float16)f2[i];
    a[i + 12] = (_Float16)f3[i];
  }
  return a;
}

// ---------------------------------------------------------------------------
// K0: weight fp32 -> f16 with transpose:  WT[n][k] = (f16) W[k][n]
// Uses gfx1250 async global->LDS copy (ASYNCcnt) to stage the 32x32 tile.
// Row stride padded to 36 floats so every 16B async chunk is 16B-aligned.
// ---------------------------------------------------------------------------
__global__ __launch_bounds__(256)
void wcvt_kernel(const float* __restrict__ W, _Float16* __restrict__ WT) {
  __shared__ float t[32][36];
  const int tid = threadIdx.y * 32 + threadIdx.x;  // 0..255
  const int row = tid >> 3;                        // 0..31
  const int ch  = tid & 7;                         // 0..7 (16B chunks of a row)
  const float* gsrc =
      W + (size_t)(blockIdx.y * 32 + row) * DM + blockIdx.x * 32 + ch * 4;
  // low 32 bits of a generic LDS pointer = LDS byte offset (flat->LDS trunc)
  unsigned int ldsdst = (unsigned int)(size_t)&t[row][ch * 4];
  asm volatile("global_load_async_to_lds_b128 %0, %1, off"
               :: "v"(ldsdst), "v"((unsigned long long)(size_t)gsrc)
               : "memory");
  asm volatile("s_wait_asynccnt 0x0" ::: "memory");
  __syncthreads();
  int tx = threadIdx.x, ty = threadIdx.y;
  int xo = blockIdx.y * 32 + tx;
#pragma unroll
  for (int i = 0; i < 4; ++i)
    WT[(size_t)(blockIdx.x * 32 + ty + 8 * i) * DM + xo] = (_Float16)t[tx][ty + 8 * i];
}

// ---------------------------------------------------------------------------
// K1: projection GEMM  Y(M=8192,N=1024) = X(f32) @ W  via WT(f16), out f16.
// Software-pipelined: next K-step fragments load while current WMMAs issue.
// transposeOut==0 -> Y[b][h][s][d]   (for Q,K)
// transposeOut==1 -> Y[b][h][d][s]   (for V, so attn@V B-loads are contiguous)
// ---------------------------------------------------------------------------
__global__ __launch_bounds__(128)
void proj_kernel(const float* __restrict__ X, const _Float16* __restrict__ WT,
                 _Float16* __restrict__ Y, int transposeOut, float scale) {
  const int lane = threadIdx.x & 31;
  const int wave = threadIdx.x >> 5;
  const int M0 = blockIdx.x * 64 + wave * 16;
  const int N0 = blockIdx.y * 64;
  const int rowA = M0 + (lane & 15);
  const float* xrow = X + (size_t)rowA * DM;
  const int ncol = lane & 15;
  const int koff = (lane & 16) ? 16 : 0;
  const _Float16* bbase = WT + (size_t)(N0 + ncol) * DM + koff;

  v8f acc[4] = {};
  v16h a_cur = load_a_f32(xrow, 0, lane);
  v16h b_cur[4];
#pragma unroll
  for (int j = 0; j < 4; ++j)
    b_cur[j] = *(const v16h*)(bbase + (size_t)j * 16 * DM);

  for (int kk = 0; kk < DM - 32; kk += 32) {
    v16h a_nxt = load_a_f32(xrow, kk + 32, lane);
    v16h b_nxt[4];
#pragma unroll
    for (int j = 0; j < 4; ++j)
      b_nxt[j] = *(const v16h*)(bbase + (kk + 32) + (size_t)j * 16 * DM);
    __builtin_prefetch(bbase + kk + 96, 0, 1);
#pragma unroll
    for (int j = 0; j < 4; ++j)
      acc[j] = wmma_f16(a_cur, b_cur[j], acc[j]);
    a_cur = a_nxt;
#pragma unroll
    for (int j = 0; j < 4; ++j) b_cur[j] = b_nxt[j];
  }
#pragma unroll
  for (int j = 0; j < 4; ++j)
    acc[j] = wmma_f16(a_cur, b_cur[j], acc[j]);

  const int hi8 = (lane & 16) ? 8 : 0;
#pragma unroll
  for (int j = 0; j < 4; ++j) {
#pragma unroll
    for (int r = 0; r < 8; ++r) {
      int m = M0 + r + hi8;
      int n = N0 + j * 16 + ncol;
      int b = m >> 10, s = m & (SEQ - 1);
      int h = n >> 6,  d = n & (DH - 1);
      size_t idx = transposeOut
                     ? ((size_t)((b * NH + h) * DH + d)) * SEQ + s
                     : ((size_t)((b * NH + h) * SEQ + s)) * DH + d;
      Y[idx] = (_Float16)(acc[j][r] * scale);
    }
  }
}

// ---------------------------------------------------------------------------
// K2: fused attention per (b,h,16-query tile), one wave32 per block.
//   pass1: scores via WMMA -> row max     (Q scale pre-folded)
//   pass2: recompute, e=exp(s-max) -> LDS f16, row sums
//   pass3: write normalized attn fp32 to d_out, renormalize LDS
//   pass4: context = attn @ V via WMMA from LDS, store ctx f16 (B,S,H*64)
// ---------------------------------------------------------------------------
__global__ __launch_bounds__(32)
void attn_kernel(const _Float16* __restrict__ Qh, const _Float16* __restrict__ Kh,
                 const _Float16* __restrict__ VT, const unsigned char* __restrict__ mask,
                 float* __restrict__ attn_out, _Float16* __restrict__ ctx) {
  __shared__ _Float16 at[16][SEQ];   // 32 KB
  __shared__ float rinvS[16];

  const int lane = threadIdx.x;
  const int qt = blockIdx.x & (SEQ / 16 - 1);
  const int bh = blockIdx.x >> 6;
  const int b = bh >> 4, h = bh & (NH - 1);
  const int q0 = qt * 16;

  const _Float16* Qbase = Qh + ((size_t)bh * SEQ + q0) * DH;
  const _Float16* Kbase = Kh + (size_t)bh * SEQ * DH;
  const _Float16* Vbase = VT + (size_t)bh * DH * SEQ;
  const unsigned char* mbase = mask + ((size_t)b * SEQ + q0) * SEQ;

  const int mrow = lane & 15;
  const int ncol = lane & 15;
  const int hi8  = (lane & 16) ? 8 : 0;
  const int dsel = (lane & 16) ? 16 : 0;

  v16h a0 = load_a_f16(Qbase + (size_t)mrow * DH, 0, lane);
  v16h a1 = load_a_f16(Qbase + (size_t)mrow * DH, 32, lane);

  float pm[8];
#pragma unroll
  for (int r = 0; r < 8; ++r) pm[r] = -3.4e38f;

  // ---- pass 1: row max of masked scaled scores ----
  for (int kt = 0; kt < SEQ / 16; ++kt) {
    int key = kt * 16 + ncol;
    const _Float16* krow = Kbase + (size_t)key * DH;
    v16h b0 = *(const v16h*)(krow + dsel);
    v16h b1 = *(const v16h*)(krow + 32 + dsel);
    v8f c = {};
    c = wmma_f16(a0, b0, c);
    c = wmma_f16(a1, b1, c);
#pragma unroll
    for (int r = 0; r < 8; ++r) {
      float sc = mbase[(size_t)(r + hi8) * SEQ + key] ? NEGBIG : c[r];
      pm[r] = fmaxf(pm[r], sc);
    }
  }
#pragma unroll
  for (int r = 0; r < 8; ++r)
#pragma unroll
    for (int m = 1; m <= 8; m <<= 1)
      pm[r] = fmaxf(pm[r], __shfl_xor(pm[r], m));

  // ---- pass 2: exp, row sums, store e (f16) to LDS ----
  float ps[8] = {0.f, 0.f, 0.f, 0.f, 0.f, 0.f, 0.f, 0.f};
  for (int kt = 0; kt < SEQ / 16; ++kt) {
    int key = kt * 16 + ncol;
    const _Float16* krow = Kbase + (size_t)key * DH;
    v16h b0 = *(const v16h*)(krow + dsel);
    v16h b1 = *(const v16h*)(krow + 32 + dsel);
    v8f c = {};
    c = wmma_f16(a0, b0, c);
    c = wmma_f16(a1, b1, c);
#pragma unroll
    for (int r = 0; r < 8; ++r) {
      int qq = r + hi8;
      float sc = mbase[(size_t)qq * SEQ + key] ? NEGBIG : c[r];
      float e = __expf(sc - pm[r]);
      ps[r] += e;
      at[qq][key] = (_Float16)e;
    }
  }
#pragma unroll
  for (int r = 0; r < 8; ++r)
#pragma unroll
    for (int m = 1; m <= 8; m <<= 1)
      ps[r] += __shfl_xor(ps[r], m);
  if (lane == 0)
#pragma unroll
    for (int r = 0; r < 8; ++r) rinvS[r] = 1.0f / ps[r];
  if (lane == 16)
#pragma unroll
    for (int r = 0; r < 8; ++r) rinvS[8 + r] = 1.0f / ps[r];
  __syncthreads();

  // ---- pass 3: write attn (fp32) coalesced, renormalize LDS copy ----
  for (int r = 0; r < 16; ++r) {
    float ri = rinvS[r];
    size_t obase = ((size_t)bh * SEQ + (q0 + r)) * SEQ;
    for (int cI = lane; cI < SEQ; cI += 32) {
      float av = (float)at[r][cI] * ri;
      attn_out[obase + cI] = av;
      at[r][cI] = (_Float16)av;
    }
  }
  __syncthreads();

  // ---- pass 4: context = attn @ V  (16 x 64), A from LDS, B from VT ----
  v8f acc[4] = {};
  for (int kk = 0; kk < SEQ; kk += 32) {
    v16h a = load_a_f16(&at[mrow][0], kk, lane);
#pragma unroll
    for (int j = 0; j < 4; ++j) {
      const _Float16* vrow = Vbase + (size_t)(j * 16 + ncol) * SEQ + kk + dsel;
      v16h bb = *(const v16h*)(vrow);
      acc[j] = wmma_f16(a, bb, acc[j]);
    }
  }
#pragma unroll
  for (int j = 0; j < 4; ++j) {
#pragma unroll
    for (int r = 0; r < 8; ++r) {
      int q = q0 + r + hi8;
      int n = h * DH + j * 16 + ncol;
      ctx[((size_t)b * SEQ + q) * DM + n] = (_Float16)acc[j][r];
    }
  }
}

// ---------------------------------------------------------------------------
// K3: raw = ctx(f16) @ W_fc (via WT f16) + Q_in (residual), fp32 out.
// Software-pipelined like K1.
// ---------------------------------------------------------------------------
__global__ __launch_bounds__(128)
void fcgemm_kernel(const _Float16* __restrict__ X, const _Float16* __restrict__ WT,
                   const float* __restrict__ resid, float* __restrict__ raw) {
  const int lane = threadIdx.x & 31;
  const int wave = threadIdx.x >> 5;
  const int M0 = blockIdx.x * 64 + wave * 16;
  const int N0 = blockIdx.y * 64;
  const int rowA = M0 + (lane & 15);
  const _Float16* xrow = X + (size_t)rowA * DM;
  const int ncol = lane & 15;
  const int koff = (lane & 16) ? 16 : 0;
  const _Float16* bbase = WT + (size_t)(N0 + ncol) * DM + koff;

  v8f acc[4] = {};
  v16h a_cur = load_a_f16(xrow, 0, lane);
  v16h b_cur[4];
#pragma unroll
  for (int j = 0; j < 4; ++j)
    b_cur[j] = *(const v16h*)(bbase + (size_t)j * 16 * DM);

  for (int kk = 0; kk < DM - 32; kk += 32) {
    v16h a_nxt = load_a_f16(xrow, kk + 32, lane);
    v16h b_nxt[4];
#pragma unroll
    for (int j = 0; j < 4; ++j)
      b_nxt[j] = *(const v16h*)(bbase + (kk + 32) + (size_t)j * 16 * DM);
    __builtin_prefetch(bbase + kk + 96, 0, 1);
#pragma unroll
    for (int j = 0; j < 4; ++j)
      acc[j] = wmma_f16(a_cur, b_cur[j], acc[j]);
    a_cur = a_nxt;
#pragma unroll
    for (int j = 0; j < 4; ++j) b_cur[j] = b_nxt[j];
  }
#pragma unroll
  for (int j = 0; j < 4; ++j)
    acc[j] = wmma_f16(a_cur, b_cur[j], acc[j]);

  const int hi8 = (lane & 16) ? 8 : 0;
#pragma unroll
  for (int j = 0; j < 4; ++j) {
#pragma unroll
    for (int r = 0; r < 8; ++r) {
      size_t m = (size_t)(M0 + r + hi8);
      int n = N0 + j * 16 + ncol;
      raw[m * DM + n] = acc[j][r] + resid[m * DM + n];
    }
  }
}

// ---------------------------------------------------------------------------
// K4: LayerNorm over last dim, one 256-thread block per row
// ---------------------------------------------------------------------------
__global__ __launch_bounds__(256)
void ln_kernel(const float* __restrict__ raw, const float* __restrict__ gamma,
               const float* __restrict__ beta, float* __restrict__ out) {
  __shared__ float s1[256], s2[256];
  const size_t row = blockIdx.x;
  const int t = threadIdx.x;
  const float* r = raw + row * DM;
  float v[4], sum = 0.f, sq = 0.f;
#pragma unroll
  for (int i = 0; i < 4; ++i) {
    v[i] = r[t + 256 * i];
    sum += v[i];
    sq += v[i] * v[i];
  }
  s1[t] = sum; s2[t] = sq;
  __syncthreads();
  for (int off = 128; off > 0; off >>= 1) {
    if (t < off) { s1[t] += s1[t + off]; s2[t] += s2[t + off]; }
    __syncthreads();
  }
  float mu = s1[0] * (1.0f / DM);
  float var = s2[0] * (1.0f / DM) - mu * mu;
  float rs = rsqrtf(var + 1e-5f);
#pragma unroll
  for (int i = 0; i < 4; ++i) {
    int c = t + 256 * i;
    out[row * DM + c] = (v[i] - mu) * rs * gamma[c] + beta[c];
  }
}

// ---------------------------------------------------------------------------
// launch
// ---------------------------------------------------------------------------
extern "C" void kernel_launch(void* const* d_in, const int* in_sizes, int n_in,
                              void* d_out, int out_size, void* d_ws, size_t ws_size,
                              hipStream_t stream) {
  const float* Q_in = (const float*)d_in[0];
  const float* K_in = (const float*)d_in[1];
  const float* V_in = (const float*)d_in[2];
  const unsigned char* mask = (const unsigned char*)d_in[3];
  const float* W_Q  = (const float*)d_in[4];
  const float* W_K  = (const float*)d_in[5];
  const float* W_V  = (const float*)d_in[6];
  const float* W_fc = (const float*)d_in[7];
  const float* ln_g = (const float*)d_in[8];
  const float* ln_b = (const float*)d_in[9];

  float* out  = (float*)d_out;
  float* attn = out + (size_t)BB * SEQ * DM;   // 8M floats of `out`, then attn

  char* ws = (char*)d_ws;
  _Float16* wqt = (_Float16*)(ws + (size_t)0);
  _Float16* wkt = (_Float16*)(ws + ((size_t)2  << 20));
  _Float16* wvt = (_Float16*)(ws + ((size_t)4  << 20));
  _Float16* wft = (_Float16*)(ws + ((size_t)6  << 20));
  _Float16* qh  = (_Float16*)(ws + ((size_t)8  << 20));   // (B,H,S,64) f16, scaled
  _Float16* kh  = (_Float16*)(ws + ((size_t)24 << 20));   // (B,H,S,64) f16
  _Float16* vth = (_Float16*)(ws + ((size_t)40 << 20));   // (B,H,64,S) f16
  _Float16* ctx = (_Float16*)(ws + ((size_t)56 << 20));   // (B,S,H*64) f16
  float*    raw = (float*)   (ws + ((size_t)72 << 20));   // (B,S,1024) f32

  dim3 tgrid(DM / 32, DM / 32), tblk(32, 8);
  wcvt_kernel<<<tgrid, tblk, 0, stream>>>(W_Q,  wqt);
  wcvt_kernel<<<tgrid, tblk, 0, stream>>>(W_K,  wkt);
  wcvt_kernel<<<tgrid, tblk, 0, stream>>>(W_V,  wvt);
  wcvt_kernel<<<tgrid, tblk, 0, stream>>>(W_fc, wft);

  dim3 ggrid((BB * SEQ) / 64, DM / 64);
  proj_kernel<<<ggrid, 128, 0, stream>>>(Q_in, wqt, qh, 0, 0.125f); // fold 1/sqrt(64)
  proj_kernel<<<ggrid, 128, 0, stream>>>(K_in, wkt, kh, 0, 1.0f);
  proj_kernel<<<ggrid, 128, 0, stream>>>(V_in, wvt, vth, 1, 1.0f);

  attn_kernel<<<BB * NH * (SEQ / 16), 32, 0, stream>>>(qh, kh, vth, mask, attn, ctx);

  fcgemm_kernel<<<ggrid, 128, 0, stream>>>(ctx, wft, Q_in, raw);

  ln_kernel<<<BB * SEQ, 256, 0, stream>>>(raw, ln_g, ln_b, out);
}
